// GraphLSTM_47536698032461
// MI455X (gfx1250) — matrix operand order repaired and checked by reading, not compiled, not run on hardware
//
#include <hip/hip_runtime.h>
#include <hip/hip_bf16.h>

// ---------------------------------------------------------------------------
// GraphLSTM on MI455X (gfx1250).
//   - 32 LSTM cell evaluations; each cell's h@Whh.T (2048x4096x1024) runs on
//     v_wmma_f32_16x16x32_bf16 (bf16 in, fp32 accumulate).
//   - Whh row-major [4H][K] is read directly as the WMMA B operand
//     (lane = N%16, 16 contiguous K per lane, same k order as A) -> cast to bf16, no repack.
//   - Per wave: 32x32 h-tile, all 4 gate quadrants accumulated in registers
//     (16 x v8f accumulators), activations fused in the epilogue.
//   - Activations kept bf16 so each cell's output feeds the next A operand.
// ---------------------------------------------------------------------------

typedef __bf16 bf16;
typedef __bf16 bf16x8  __attribute__((ext_vector_type(8)));
typedef __bf16 bf16x16 __attribute__((ext_vector_type(16)));
typedef float  f32x8   __attribute__((ext_vector_type(8)));

#define B_DIM 2048
#define H_DIM 1024
#define G4H   4096

// ---------------- Whh fp32 -> bf16 cast (layout unchanged) -----------------
__global__ __launch_bounds__(256) void glstm_convert_whh(
    const float* __restrict__ whh, bf16* __restrict__ out, int n) {
  int i = blockIdx.x * 256 + threadIdx.x;
  if (i < n) out[i] = (bf16)whh[i];
}

// ---------------- tree-join: average 2 or 3 (h, c) states ------------------
__global__ __launch_bounds__(256) void glstm_merge(
    const bf16* __restrict__ h0, const bf16* __restrict__ h1, const bf16* h2,
    const float* __restrict__ c0, const float* __restrict__ c1, const float* c2,
    bf16* __restrict__ ho, float* __restrict__ co, float inv, int n) {
  int i = blockIdx.x * 256 + threadIdx.x;
  if (i >= n) return;
  float h = (float)h0[i] + (float)h1[i] + (h2 ? (float)h2[i] : 0.0f);
  float c = c0[i] + c1[i] + (c2 ? c2[i] : 0.0f);
  ho[i] = (bf16)(h * inv);
  co[i] = c * inv;
}

__device__ __forceinline__ float sigmoidf_(float x) {
  return 1.0f / (1.0f + __expf(-x));
}

// ---------------- fused LSTM cell: WMMA matmul + gate activations ----------
// Grid: (B/64, H/128); block = 256 threads = 8 waves; wave tile = 32x32 of h.
__global__ __launch_bounds__(256) void glstm_cell(
    const bf16*  __restrict__ h_in,   // [B][H] bf16, nullptr => zero state
    const float* __restrict__ c_in,   // [B][H] f32,  nullptr => zero state
    const float* __restrict__ p3d,    // [B][16][3]
    int joint,
    const float* __restrict__ Wih,    // [4H][3]
    const float* __restrict__ bih,    // [4H]
    const float* __restrict__ bhh,    // [4H]
    const bf16*  __restrict__ whh,    // [4H][H] bf16 row-major (B operand)
    bf16*  __restrict__ h_out,        // [B][H] bf16
    float* __restrict__ c_out)        // [B][H] f32
{
  const int lane = threadIdx.x & 31;
  const int wave = threadIdx.x >> 5;
  const int l16  = lane & 15;
  const int lh   = lane >> 4;  // lane half: 0 or 1

  const int m0 = blockIdx.x * 64  + (wave & 1) * 32;   // batch rows
  const int n0 = blockIdx.y * 128 + (wave >> 1) * 32;  // h columns

  // 2x2 output tiles x 4 gates, fp32 accumulators
  f32x8 acc[2][2][4];
  #pragma unroll
  for (int mi = 0; mi < 2; ++mi)
    #pragma unroll
    for (int nj = 0; nj < 2; ++nj)
      #pragma unroll
      for (int g = 0; g < 4; ++g) {
        f32x8 z = {};
        acc[mi][nj][g] = z;
      }

  if (h_in) {
    #pragma unroll 1
    for (int k0 = 0; k0 < H_DIM; k0 += 32) {
      // A fragments: lane holds row m0+mi*16+l16, K = k0+lh*16..+15
      bf16x16 a[2];
      #pragma unroll
      for (int mi = 0; mi < 2; ++mi) {
        const bf16* ap = h_in + (size_t)(m0 + mi * 16 + l16) * H_DIM + k0 + lh * 16;
        bf16x8 lo = *(const bf16x8*)(ap);
        bf16x8 hi = *(const bf16x8*)(ap + 8);
        #pragma unroll
        for (int e = 0; e < 8; ++e) { a[mi][e] = lo[e]; a[mi][8 + e] = hi[e]; }
      }
      // B fragments: lane holds col g*H + n0+nj*16+l16, K = k0+lh*16..+15
      bf16x16 bm[4][2];
      #pragma unroll
      for (int g = 0; g < 4; ++g)
        #pragma unroll
        for (int nj = 0; nj < 2; ++nj) {
          const bf16* bp = whh + (size_t)(g * H_DIM + n0 + nj * 16 + l16) * H_DIM
                               + k0 + lh * 16;
          bf16x8 lo = *(const bf16x8*)(bp);
          bf16x8 hi = *(const bf16x8*)(bp + 8);
          #pragma unroll
          for (int e = 0; e < 8; ++e) { bm[g][nj][e] = lo[e]; bm[g][nj][8 + e] = hi[e]; }
        }
      // 16 WMMAs per K-step
      #pragma unroll
      for (int mi = 0; mi < 2; ++mi)
        #pragma unroll
        for (int nj = 0; nj < 2; ++nj)
          #pragma unroll
          for (int g = 0; g < 4; ++g)
            acc[mi][nj][g] = __builtin_amdgcn_wmma_f32_16x16x32_bf16(
                false, a[mi], false, bm[g][nj], (short)0, acc[mi][nj][g],
                false, false);
    }
  }

  // Epilogue: fold in x@Wih.T + biases, then gate activations.
  float wx[4][2][3], bsum[4][2];
  #pragma unroll
  for (int g = 0; g < 4; ++g)
    #pragma unroll
    for (int nj = 0; nj < 2; ++nj) {
      int col = g * H_DIM + n0 + nj * 16 + l16;
      wx[g][nj][0] = Wih[col * 3 + 0];
      wx[g][nj][1] = Wih[col * 3 + 1];
      wx[g][nj][2] = Wih[col * 3 + 2];
      bsum[g][nj]  = bih[col] + bhh[col];
    }

  #pragma unroll
  for (int mi = 0; mi < 2; ++mi) {
    #pragma unroll
    for (int e = 0; e < 8; ++e) {
      int R = m0 + mi * 16 + lh * 8 + e;  // C/D layout: elem e -> row (+8 for hi half)
      float x0 = p3d[R * 48 + joint * 3 + 0];
      float x1 = p3d[R * 48 + joint * 3 + 1];
      float x2 = p3d[R * 48 + joint * 3 + 2];
      #pragma unroll
      for (int nj = 0; nj < 2; ++nj) {
        int C = n0 + nj * 16 + l16;
        float gv[4];
        #pragma unroll
        for (int g = 0; g < 4; ++g)
          gv[g] = acc[mi][nj][g][e] + x0 * wx[g][nj][0] + x1 * wx[g][nj][1]
                + x2 * wx[g][nj][2] + bsum[g][nj];
        float ig = sigmoidf_(gv[0]);          // torch gate order i,f,g,o
        float fg = sigmoidf_(gv[1]);
        float gg = tanhf(gv[2]);
        float og = sigmoidf_(gv[3]);
        float cp = c_in ? c_in[(size_t)R * H_DIM + C] : 0.0f;
        float c2 = fg * cp + ig * gg;
        float h  = og * tanhf(c2);
        c_out[(size_t)R * H_DIM + C] = c2;
        h_out[(size_t)R * H_DIM + C] = (bf16)h;
      }
    }
  }
}

// ---------------- final projection: ((fh+bh)/2) @ Wref.T + bref ------------
// One wave per (batch, joint) pair; lane-parallel K with shuffle reduction.
__global__ __launch_bounds__(256) void glstm_proj(
    const bf16* __restrict__ fh, const bf16* __restrict__ bh,
    const float* __restrict__ Wref, const float* __restrict__ bref,
    float* __restrict__ out) {
  int lane = threadIdx.x & 31, wave = threadIdx.x >> 5;
  int p = blockIdx.x * 8 + wave;  // 0..B*16-1
  int b = p >> 4, j = p & 15;
  const bf16* f = fh + ((size_t)j * B_DIM + b) * H_DIM;
  const bf16* r = bh + ((size_t)j * B_DIM + b) * H_DIM;
  float a0 = 0.f, a1 = 0.f, a2 = 0.f;
  for (int k = lane; k < H_DIM; k += 32) {
    float h = 0.5f * ((float)f[k] + (float)r[k]);
    a0 += h * Wref[k];
    a1 += h * Wref[H_DIM + k];
    a2 += h * Wref[2 * H_DIM + k];
  }
  #pragma unroll
  for (int off = 16; off; off >>= 1) {
    a0 += __shfl_down(a0, off);
    a1 += __shfl_down(a1, off);
    a2 += __shfl_down(a2, off);
  }
  if (lane == 0) {
    float* o = out + ((size_t)b * 16 + j) * 3;
    o[0] = a0 + bref[0];
    o[1] = a1 + bref[1];
    o[2] = a2 + bref[2];
  }
}

// ---------------------------------------------------------------------------
extern "C" void kernel_launch(void* const* d_in, const int* in_sizes, int n_in,
                              void* d_out, int out_size, void* d_ws, size_t ws_size,
                              hipStream_t stream) {
  const float* p3d  = (const float*)d_in[0];
  const float* Wih  = (const float*)d_in[1];
  const float* Whh  = (const float*)d_in[2];
  const float* bih  = (const float*)d_in[3];
  const float* bhh  = (const float*)d_in[4];
  const float* Wref = (const float*)d_in[5];
  const float* bref = (const float*)d_in[6];
  float* out = (float*)d_out;

  // Workspace layout (~276 MB):
  //   whh bf16 8MB | fh[16] 64MB | bh[16] 64MB | c[16] 128MB | hx 4MB | cx 8MB
  const size_t S_h = (size_t)B_DIM * H_DIM * sizeof(bf16);
  const size_t S_c = (size_t)B_DIM * H_DIM * sizeof(float);
  char* ws = (char*)d_ws;
  bf16*  whh_bf = (bf16*)ws;  ws += (size_t)G4H * H_DIM * sizeof(bf16);
  bf16*  fhb    = (bf16*)ws;  ws += 16 * S_h;
  bf16*  bhb    = (bf16*)ws;  ws += 16 * S_h;
  float* cbuf   = (float*)ws; ws += 16 * S_c;
  bf16*  hx     = (bf16*)ws;  ws += S_h;
  float* cx     = (float*)ws; ws += S_c;

  auto FH = [&](int j) { return fhb  + (size_t)j * B_DIM * H_DIM; };
  auto BH = [&](int j) { return bhb  + (size_t)j * B_DIM * H_DIM; };
  auto CC = [&](int j) { return cbuf + (size_t)j * B_DIM * H_DIM; };

  {  // cast Whh once per launch (deterministic)
    int n = G4H * H_DIM;
    glstm_convert_whh<<<(n + 255) / 256, 256, 0, stream>>>(Whh, whh_bf, n);
  }

  dim3 cg(B_DIM / 64, H_DIM / 128);  // (32, 8) blocks, 256 threads each
  auto cell = [&](int joint, const bf16* hi, const float* ci, bf16* ho, float* co) {
    glstm_cell<<<cg, 256, 0, stream>>>(hi, ci, p3d, joint, Wih, bih, bhh,
                                       whh_bf, ho, co);
  };
  const int nE = B_DIM * H_DIM;
  auto merge2 = [&](const bf16* h0, const bf16* h1, const float* c0, const float* c1) {
    glstm_merge<<<(nE + 255) / 256, 256, 0, stream>>>(h0, h1, nullptr, c0, c1,
                                                      nullptr, hx, cx, 0.5f, nE);
  };
  auto merge3 = [&](const bf16* h0, const bf16* h1, const bf16* h2,
                    const float* c0, const float* c1, const float* c2) {
    glstm_merge<<<(nE + 255) / 256, 256, 0, stream>>>(h0, h1, h2, c0, c1, c2,
                                                      hx, cx, 1.0f / 3.0f, nE);
  };

  static const int chains[4][3] = {{15, 14, 13}, {10, 11, 12}, {5, 4, 3}, {0, 1, 2}};

  // -------- forward pass --------
  for (int c = 0; c < 4; ++c) {
    const int* ch = chains[c];
    cell(ch[0], nullptr,   nullptr,    FH(ch[0]), CC(ch[0]));
    cell(ch[1], FH(ch[0]), CC(ch[0]),  FH(ch[1]), CC(ch[1]));
    cell(ch[2], FH(ch[1]), CC(ch[1]),  FH(ch[2]), CC(ch[2]));
  }
  merge2(FH(2), FH(3), CC(2), CC(3));
  cell(6, hx, cx, FH(6), CC(6));
  cell(7, FH(6), CC(6), FH(7), CC(7));
  merge3(FH(7), FH(13), FH(12), CC(7), CC(13), CC(12));
  cell(8, hx, cx, FH(8), CC(8));
  cell(9, FH(8), CC(8), FH(9), CC(9));

  // -------- backward pass (forward c-slots no longer needed; reuse) --------
  cell(9, nullptr, nullptr, BH(9), CC(9));
  cell(8, BH(9), CC(9), BH(8), CC(8));
  cell(7, BH(8), CC(8), BH(7), CC(7));
  cell(6, BH(7), CC(7), BH(6), CC(6));
  for (int c = 0; c < 4; ++c) {
    const int* ch = chains[c];
    int s = (ch[2] > 9) ? 8 : 6;
    cell(ch[2], BH(s),     CC(s),      BH(ch[2]), CC(ch[2]));
    cell(ch[1], BH(ch[2]), CC(ch[2]),  BH(ch[1]), CC(ch[1]));
    cell(ch[0], BH(ch[1]), CC(ch[1]),  BH(ch[0]), CC(ch[0]));
  }

  // -------- output projection --------
  glstm_proj<<<(B_DIM * 16) / 8, 256, 0, stream>>>(fhb, bhb, Wref, bref, out);
}
